// Decoder_77953656422783
// MI455X (gfx1250) — compile-verified, matrix operand order checked by reading, not run on hardware
//
#include <hip/hip_runtime.h>

// ---------------------------------------------------------------------------
// Decoder: attention-LSTM, restructured as
//   prep (bf16 converts, W_cat=[W_ih|W_hh], X gather, UV GEMM)
//   31 x { attn (VALU) -> gates WMMA -> LSTM pointwise -> proj WMMA }
//   one big logits WMMA GEMM (1984 x 30000 x 256), 64-row A supertile in LDS
// All GEMMs: v_wmma_f32_16x16x32_bf16, wave32, fp32 accumulate.
// ---------------------------------------------------------------------------

#define VOCAB 30000
#define EMB   256
#define HDIM  512
#define VDIM  128
#define ATT   256
#define BSZ   64
#define NN    196
#define TT    32
#define TS    31   // T-1 decode steps

typedef __attribute__((ext_vector_type(16))) __bf16 v16bf;
typedef __attribute__((ext_vector_type(8)))  float  v8f;
typedef int v4i __attribute__((ext_vector_type(4)));
typedef __attribute__((address_space(1))) v4i g_v4i;   // global int4
typedef __attribute__((address_space(3))) v4i l_v4i;   // LDS int4

#if defined(__has_builtin)
#if __has_builtin(__builtin_amdgcn_global_load_async_to_lds_b128)
#define HAVE_ASYNC_LDS 1
#endif
#endif

__device__ __forceinline__ unsigned short f2bf(float x) {
  union { float f; unsigned u; } v; v.f = x;
  unsigned r = v.u + 0x7FFFu + ((v.u >> 16) & 1u);   // round-nearest-even
  return (unsigned short)(r >> 16);
}

// Load one wave's A (or B) fragment for wmma_f32_16x16x32_bf16.
// ISA layout: lane L holds row (L&15); VGPR halves cover
//   k = k0 + 8*(L>>4) + {0..7}  and  k = k0 + 16 + 8*(L>>4) + {0..7}
__device__ __forceinline__ v16bf load_frag(const unsigned short* base,
                                           int row, int stride, int k0, int lane) {
  int khalf = lane >> 4;
  size_t off = (size_t)row * stride + k0 + khalf * 8;
  union { uint4 u[2]; v16bf v; } u;
  u.u[0] = *(const uint4*)(base + off);
  u.u[1] = *(const uint4*)(base + off + 16);
  return u.v;
}

__device__ __forceinline__ v16bf load_frag_lds(const unsigned short* lds,
                                               int lane, int k0) {
  int m = lane & 15, khalf = lane >> 4;
  int off = m * 256 + k0 + khalf * 8;
  union { uint4 u[2]; v16bf v; } u;
  u.u[0] = *(const uint4*)(lds + off);
  u.u[1] = *(const uint4*)(lds + off + 16);
  return u.v;
}

__device__ __forceinline__ v8f wmma_bf16(v16bf a, v16bf b, v8f c) {
  return __builtin_amdgcn_wmma_f32_16x16x32_bf16(false, a, false, b,
                                                 (short)0, c, false, false);
}

// ------------------------------ prep kernels -------------------------------

__global__ void k_f2bf(const float* __restrict__ src,
                       unsigned short* __restrict__ dst, int n) {
  int i = blockIdx.x * 256 + threadIdx.x;
  if (i < n) dst[i] = f2bf(src[i]);
}

// W_cat[j, 0:384] = W_ih[j,:], W_cat[j, 384:896] = W_hh[j,:]   (bf16)
__global__ void k_wcat(const float* __restrict__ Wih, const float* __restrict__ Whh,
                       unsigned short* __restrict__ Wcat) {
  int i = blockIdx.x * 256 + threadIdx.x;          // 2048*896
  if (i >= 2048 * 896) return;
  int j = i / 896, k = i % 896;
  float v = (k < 384) ? Wih[j * 384 + k] : Whh[j * 512 + (k - 384)];
  Wcat[i] = f2bf(v);
}

// Xb[(t*64+b)*256 + e] = bf16(embed[y[b,t], e])  for t = 0..30
__global__ void k_gatherx(const int* __restrict__ y, const float* __restrict__ embed,
                          unsigned short* __restrict__ Xb) {
  int i = blockIdx.x * 256 + threadIdx.x;          // 31*64*256
  if (i >= TS * BSZ * EMB) return;
  int t = i / (BSZ * EMB);
  int b = (i / EMB) % BSZ;
  int e = i % EMB;
  int tok = y[b * TT + t];
  Xb[i] = f2bf(embed[(size_t)tok * EMB + e]);
}

__global__ void k_init_state(float* __restrict__ h, float* __restrict__ c,
                             unsigned short* __restrict__ hb) {
  int i = blockIdx.x * 256 + threadIdx.x;          // 64*512
  if (i < BSZ * HDIM) { h[i] = 0.f; c[i] = 0.f; hb[i] = 0; }
}

// ----------------------- UV = V @ U^T + b  (WMMA) --------------------------
// M = 12544 (784 tiles), N = 256 (16 tiles), K = 128 (4 k-steps)
__global__ void k_uv(const unsigned short* __restrict__ Vb,
                     const unsigned short* __restrict__ Ub,
                     const float* __restrict__ ubias,
                     float* __restrict__ UV) {
  int lane = threadIdx.x & 31, wave = threadIdx.x >> 5;
  int mtile = blockIdx.x;
  int ntile = blockIdx.y * 8 + wave;
  int arow = mtile * 16 + (lane & 15);
  int brow = ntile * 16 + (lane & 15);
  v8f acc = {};
  for (int ks = 0; ks < 4; ks++) {
    v16bf a = load_frag(Vb, arow, VDIM, ks * 32, lane);
    v16bf b = load_frag(Ub, brow, VDIM, ks * 32, lane);
    acc = wmma_bf16(a, b, acc);
  }
  int n = ntile * 16 + (lane & 15);
  int r0 = mtile * 16 + ((lane >> 4) << 3);
  float bias = ubias[n];
  for (int j = 0; j < 8; j++)
    UV[(size_t)(r0 + j) * ATT + n] = acc[j] + bias;
}

// ------------------------- attention (per step) ----------------------------
__global__ void k_attn(const float* __restrict__ h,
                       const float* __restrict__ Ww, const float* __restrict__ Wb,
                       const float* __restrict__ UV,
                       const float* __restrict__ av, const float* __restrict__ avb,
                       const float* __restrict__ V,
                       unsigned short* __restrict__ ctxb) {
  int b = blockIdx.x, tid = threadIdx.x;
  int lane = tid & 31, wave = tid >> 5;
  __shared__ float hs[HDIM];
  __shared__ float Wh[ATT];
  __shared__ float e[224];
  __shared__ float red[8];

  hs[tid] = h[b * HDIM + tid];
  hs[tid + 256] = h[b * HDIM + 256 + tid];
  __syncthreads();

  float s = Wb[tid];
  const float4* w4 = (const float4*)(Ww + (size_t)tid * HDIM);
  const float4* h4 = (const float4*)hs;
  for (int k = 0; k < HDIM / 4; k++) {
    float4 w = w4[k], hv = h4[k];
    s += w.x * hv.x + w.y * hv.y + w.z * hv.z + w.w * hv.w;
  }
  Wh[tid] = s;
  __syncthreads();

  float vb = avb[0];
  for (int n = wave; n < NN; n += 8) {
    const float* uvrow = UV + ((size_t)b * NN + n) * ATT;
    float acc = 0.f;
    for (int a = lane; a < ATT; a += 32)
      acc += tanhf(Wh[a] + uvrow[a]) * av[a];
    for (int o = 16; o; o >>= 1) acc += __shfl_xor(acc, o, 32);
    if (lane == 0) e[n] = acc + vb;
  }
  __syncthreads();

  float mx = -1e30f;
  for (int n = tid; n < NN; n += 256) mx = fmaxf(mx, e[n]);
  for (int o = 16; o; o >>= 1) mx = fmaxf(mx, __shfl_xor(mx, o, 32));
  if (lane == 0) red[wave] = mx;
  __syncthreads();
  mx = red[0];
  for (int w = 1; w < 8; w++) mx = fmaxf(mx, red[w]);
  __syncthreads();

  float ssum = 0.f;
  for (int n = tid; n < NN; n += 256) {
    float p = __expf(e[n] - mx);
    e[n] = p;
    ssum += p;
  }
  for (int o = 16; o; o >>= 1) ssum += __shfl_xor(ssum, o, 32);
  if (lane == 0) red[wave] = ssum;
  __syncthreads();
  ssum = 0.f;
  for (int w = 0; w < 8; w++) ssum += red[w];
  float inv = 1.f / ssum;
  __syncthreads();

  if (tid < VDIM) {
    float acc = 0.f;
    for (int n = 0; n < NN; n++)
      acc += e[n] * V[((size_t)b * NN + n) * VDIM + tid];
    ctxb[b * VDIM + tid] = f2bf(acc * inv);
  }
}

// ---------------- gates = [x|ctx|h] @ [W_ih|W_hh]^T + biases ---------------
__global__ void k_gates(const unsigned short* __restrict__ Xb,
                        const unsigned short* __restrict__ ctxb,
                        const unsigned short* __restrict__ hb,
                        const unsigned short* __restrict__ Wcat,
                        const float* __restrict__ bih, const float* __restrict__ bhh,
                        float* __restrict__ gates, int t) {
  int lane = threadIdx.x & 31, wave = threadIdx.x >> 5;
  int tile = blockIdx.x * 8 + wave;                // 0..511
  int ntile = tile & 127, mtile = tile >> 7;
  const unsigned short* Xt = Xb + (size_t)t * BSZ * EMB;
  int am = mtile * 16 + (lane & 15);
  int bn = ntile * 16 + (lane & 15);
  v8f acc = {};
  for (int ks = 0; ks < 28; ks++) {
    int k0 = ks * 32;
    const unsigned short* abase; int astr, ak;
    if (k0 < 256)      { abase = Xt;   astr = 256; ak = k0; }
    else if (k0 < 384) { abase = ctxb; astr = 128; ak = k0 - 256; }
    else               { abase = hb;   astr = 512; ak = k0 - 384; }
    v16bf a = load_frag(abase, am, astr, ak, lane);
    v16bf b = load_frag(Wcat, bn, 896, k0, lane);
    acc = wmma_bf16(a, b, acc);
  }
  int n = ntile * 16 + (lane & 15);
  int r0 = mtile * 16 + ((lane >> 4) << 3);
  float bias = bih[n] + bhh[n];
  for (int j = 0; j < 8; j++)
    gates[(size_t)(r0 + j) * 2048 + n] = acc[j] + bias;
}

// ------------------------------ LSTM pointwise -----------------------------
__global__ void k_lstm(const float* __restrict__ gates,
                       float* __restrict__ c, float* __restrict__ h,
                       unsigned short* __restrict__ hb) {
  int idx = blockIdx.x * 256 + threadIdx.x;        // 64*512
  if (idx >= BSZ * HDIM) return;
  int b = idx >> 9, j = idx & 511;
  const float* g = gates + (size_t)b * 2048;
  float gi = 1.f / (1.f + __expf(-g[j]));
  float gf = 1.f / (1.f + __expf(-g[512 + j]));
  float gg = tanhf(g[1024 + j]);
  float go = 1.f / (1.f + __expf(-g[1536 + j]));
  float cn = gf * c[idx] + gi * gg;
  float hn = go * tanhf(cn);
  c[idx] = cn; h[idx] = hn; hb[idx] = f2bf(hn);
}

// ---------------------- e_t = h_new @ proj^T (WMMA) ------------------------
__global__ void k_proj(const unsigned short* __restrict__ hb,
                       const unsigned short* __restrict__ projb,
                       unsigned short* __restrict__ ebf, int t) {
  int lane = threadIdx.x & 31, wave = threadIdx.x >> 5;
  int tile = blockIdx.x * 8 + wave;                // 0..63
  int ntile = tile & 15, mtile = tile >> 4;
  int am = mtile * 16 + (lane & 15);
  int bn = ntile * 16 + (lane & 15);
  v8f acc = {};
  for (int ks = 0; ks < 16; ks++) {
    v16bf a = load_frag(hb, am, HDIM, ks * 32, lane);
    v16bf b = load_frag(projb, bn, HDIM, ks * 32, lane);
    acc = wmma_bf16(a, b, acc);
  }
  int n = ntile * 16 + (lane & 15);
  int r0 = mtile * 16 + ((lane >> 4) << 3);
  for (int j = 0; j < 8; j++)
    ebf[((size_t)t * BSZ + r0 + j) * EMB + n] = f2bf(acc[j]);
}

// -------------------- logits = E @ embed^T (big WMMA) ----------------------
// M = 1984, N = 30000, K = 256. Block: 64-row A supertile (32 KB) in LDS via
// gfx1250 async-to-LDS (ASYNCcnt); each wave holds one B fragment per k-step
// and reuses it across 4 stacked M-tiles -> 4x less embed L2 traffic.
// Each 64-row group is exactly one timestep t = blockIdx.y.
__global__ void k_logits(const unsigned short* __restrict__ E,
                         const unsigned short* __restrict__ Emb,
                         float* __restrict__ out) {
  __shared__ unsigned short As[64 * 256];          // 32 KB
  int tid = threadIdx.x, lane = tid & 31, wave = tid >> 5;
  int t = blockIdx.y;                              // 0..30 (64-row supertile)
  int ntile = blockIdx.x * 8 + wave;

  __builtin_prefetch(Emb + (size_t)blockIdx.x * 8 * 16 * EMB, 0, 0);

  // stage 64x256 bf16 A supertile: 2048 uint4, 8 per thread
  const uint4* src = (const uint4*)(E + (size_t)t * 64 * EMB);
  uint4* dst = (uint4*)As;
#if HAVE_ASYNC_LDS
  for (int i = 0; i < 8; i++) {
    __builtin_amdgcn_global_load_async_to_lds_b128(
        (g_v4i*)(unsigned long long)(src + tid + i * 256),
        (l_v4i*)(dst + tid + i * 256),
        0, 0);
  }
#if __has_builtin(__builtin_amdgcn_s_wait_asynccnt)
  __builtin_amdgcn_s_wait_asynccnt(0);
#else
  asm volatile("s_wait_asynccnt 0x0" ::: "memory");
#endif
#else
  for (int i = 0; i < 8; i++)
    dst[tid + i * 256] = src[tid + i * 256];
#endif
  __syncthreads();

  if (ntile < VOCAB / 16) {
    int bn = ntile * 16 + (lane & 15);
    v8f acc0 = {}, acc1 = {}, acc2 = {}, acc3 = {};
    for (int ks = 0; ks < 8; ks++) {
      int k0 = ks * 32;
      v16bf b = load_frag(Emb, bn, EMB, k0, lane);
      v16bf a0 = load_frag_lds(As +  0 * EMB, lane, k0);
      v16bf a1 = load_frag_lds(As + 16 * EMB, lane, k0);
      v16bf a2 = load_frag_lds(As + 32 * EMB, lane, k0);
      v16bf a3 = load_frag_lds(As + 48 * EMB, lane, k0);
      acc0 = wmma_bf16(a0, b, acc0);
      acc1 = wmma_bf16(a1, b, acc1);
      acc2 = wmma_bf16(a2, b, acc2);
      acc3 = wmma_bf16(a3, b, acc3);
    }
    // rows r = t*64 + b ; b = mm*16 + 8*(lane>>4) + j
    int bbase = (lane >> 4) << 3;
    v8f accs[4] = {acc0, acc1, acc2, acc3};
    for (int mm = 0; mm < 4; mm++) {
      for (int j = 0; j < 8; j++) {
        int b = mm * 16 + bbase + j;
        out[((size_t)b * TS + t) * VOCAB + bn] = accs[mm][j];
      }
    }
  }
}

// ------------------------------- host side ---------------------------------

static constexpr size_t alup(size_t x) { return (x + 255) & ~(size_t)255; }
static constexpr size_t SZ_EMBB = (size_t)VOCAB * EMB * 2;
static constexpr size_t SZ_WCAT = (size_t)2048 * 896 * 2;
static constexpr size_t SZ_VB   = (size_t)BSZ * NN * VDIM * 2;
static constexpr size_t SZ_UB   = (size_t)ATT * VDIM * 2;
static constexpr size_t SZ_XB   = (size_t)TS * BSZ * EMB * 2;
static constexpr size_t SZ_PRJB = (size_t)EMB * HDIM * 2;
static constexpr size_t SZ_UV   = (size_t)BSZ * NN * ATT * 4;
static constexpr size_t SZ_H    = (size_t)BSZ * HDIM * 4;
static constexpr size_t SZ_HB   = (size_t)BSZ * HDIM * 2;
static constexpr size_t SZ_CTXB = (size_t)BSZ * VDIM * 2;
static constexpr size_t SZ_GATE = (size_t)BSZ * 2048 * 4;
static constexpr size_t SZ_EBF  = (size_t)TS * BSZ * EMB * 2;

static constexpr size_t OFF_EMBB = 0;
static constexpr size_t OFF_WCAT = alup(OFF_EMBB + SZ_EMBB);
static constexpr size_t OFF_VB   = alup(OFF_WCAT + SZ_WCAT);
static constexpr size_t OFF_UB   = alup(OFF_VB   + SZ_VB);
static constexpr size_t OFF_XB   = alup(OFF_UB   + SZ_UB);
static constexpr size_t OFF_PRJB = alup(OFF_XB   + SZ_XB);
static constexpr size_t OFF_UV   = alup(OFF_PRJB + SZ_PRJB);
static constexpr size_t OFF_H    = alup(OFF_UV   + SZ_UV);
static constexpr size_t OFF_C    = alup(OFF_H    + SZ_H);
static constexpr size_t OFF_HB   = alup(OFF_C    + SZ_H);
static constexpr size_t OFF_CTXB = alup(OFF_HB   + SZ_HB);
static constexpr size_t OFF_GATE = alup(OFF_CTXB + SZ_CTXB);
static constexpr size_t OFF_EBF  = alup(OFF_GATE + SZ_GATE);
// total ~38.3 MB of workspace

extern "C" void kernel_launch(void* const* d_in, const int* in_sizes, int n_in,
                              void* d_out, int out_size, void* d_ws, size_t ws_size,
                              hipStream_t stream) {
  const float* V     = (const float*)d_in[0];
  const int*   y     = (const int*)  d_in[1];
  const float* embed = (const float*)d_in[2];
  const float* attWw = (const float*)d_in[3];
  const float* attWb = (const float*)d_in[4];
  const float* attUw = (const float*)d_in[5];
  const float* attUb = (const float*)d_in[6];
  const float* attv  = (const float*)d_in[7];
  const float* attvb = (const float*)d_in[8];
  const float* Wih   = (const float*)d_in[9];
  const float* Whh   = (const float*)d_in[10];
  const float* bih   = (const float*)d_in[11];
  const float* bhh   = (const float*)d_in[12];
  const float* projw = (const float*)d_in[13];
  float* out = (float*)d_out;
  char*  ws  = (char*)d_ws;

  unsigned short* embb  = (unsigned short*)(ws + OFF_EMBB);
  unsigned short* wcat  = (unsigned short*)(ws + OFF_WCAT);
  unsigned short* vb    = (unsigned short*)(ws + OFF_VB);
  unsigned short* ub    = (unsigned short*)(ws + OFF_UB);
  unsigned short* xb    = (unsigned short*)(ws + OFF_XB);
  unsigned short* prjb  = (unsigned short*)(ws + OFF_PRJB);
  float*          uv    = (float*)(ws + OFF_UV);
  float*          h     = (float*)(ws + OFF_H);
  float*          c     = (float*)(ws + OFF_C);
  unsigned short* hb    = (unsigned short*)(ws + OFF_HB);
  unsigned short* ctxb  = (unsigned short*)(ws + OFF_CTXB);
  float*          gates = (float*)(ws + OFF_GATE);
  unsigned short* ebf   = (unsigned short*)(ws + OFF_EBF);

  // ---- prep ----
  k_f2bf<<<(VOCAB * EMB + 255) / 256, 256, 0, stream>>>(embed, embb, VOCAB * EMB);
  k_f2bf<<<(BSZ * NN * VDIM + 255) / 256, 256, 0, stream>>>(V, vb, BSZ * NN * VDIM);
  k_f2bf<<<(ATT * VDIM + 255) / 256, 256, 0, stream>>>(attUw, ub, ATT * VDIM);
  k_f2bf<<<(EMB * HDIM + 255) / 256, 256, 0, stream>>>(projw, prjb, EMB * HDIM);
  k_wcat<<<(2048 * 896 + 255) / 256, 256, 0, stream>>>(Wih, Whh, wcat);
  k_gatherx<<<(TS * BSZ * EMB + 255) / 256, 256, 0, stream>>>(y, embed, xb);
  k_uv<<<dim3(BSZ * NN / 16, 2), 256, 0, stream>>>(vb, ub, attUb, uv);
  k_init_state<<<(BSZ * HDIM + 255) / 256, 256, 0, stream>>>(h, c, hb);

  // ---- 31 sequential decode steps (small kernels; h,c carried) ----
  for (int t = 0; t < TS; t++) {
    k_attn<<<BSZ, 256, 0, stream>>>(h, attWw, attWb, uv, attv, attvb, V, ctxb);
    k_gates<<<64, 256, 0, stream>>>(xb, ctxb, hb, wcat, bih, bhh, gates, t);
    k_lstm<<<(BSZ * HDIM + 255) / 256, 256, 0, stream>>>(gates, c, h, hb);
    k_proj<<<8, 256, 0, stream>>>(hb, prjb, ebf, t);
  }

  // ---- one big vocab GEMM for all 31 steps ----
  k_logits<<<dim3((VOCAB / 16 + 7) / 8, TS), 256, 0, stream>>>(ebf, embb, out);
}
